// SegmentLinearEmbed_20014547599736
// MI455X (gfx1250) — matrix operand order repaired and checked by reading, not compiled, not run on hardware
//
#include <hip/hip_runtime.h>

// Problem constants (from reference)
#define BB      128
#define TT      720
#define CC      862
#define SEG     48
#define NSEG    15
#define OO      256
#define KK      (CC * TT)        // 620640
#define KC      96               // K-chunk per staging iteration (620640 % 96 == 0)
#define NC      (KK / KC)        // 6465 chunks
#define NBLK    4                // grid.x: N blocks of 64 columns
#define KSPLIT  128              // grid.y: split-K factor
#define STRA    104              // LDS A row stride (halves), padded
#define STRW    112              // LDS W row stride (halves), padded (keeps 32B align)

typedef _Float16 h8  __attribute__((ext_vector_type(8)));
typedef _Float16 h16 __attribute__((ext_vector_type(16)));
typedef float    f8  __attribute__((ext_vector_type(8)));

// Exact parameter type of the async-LDS builtins (per hipcc diagnostic):
// pointer to GCC-vector 'int __attribute__((vector_size(16)))'.
typedef int v4i_vs __attribute__((vector_size(16)));
typedef __attribute__((address_space(1))) v4i_vs* gptr_v4i;
typedef __attribute__((address_space(3))) v4i_vs* lptr_v4i;

#if __has_builtin(__builtin_amdgcn_global_load_async_to_lds_b128) && \
    __has_builtin(__builtin_amdgcn_s_wait_asynccnt)
#define USE_ASYNC_LDS 1
#else
#define USE_ASYNC_LDS 0
#endif

static __device__ __forceinline__ void copy16_to_lds(const void* g, void* l) {
#if USE_ASYNC_LDS
    // gfx1250 GLOBAL_LOAD_ASYNC_TO_LDS_B128: memory -> LDS without VGPR relay,
    // tracked on ASYNCcnt.
    __builtin_amdgcn_global_load_async_to_lds_b128(
        (gptr_v4i)(g), (lptr_v4i)(l), 0, 0);
#else
    *(uint4*)l = *(const uint4*)g;
#endif
}

static __device__ __forceinline__ void wait_async_copies() {
#if USE_ASYNC_LDS
    __builtin_amdgcn_s_wait_asynccnt(0);
#endif
}

static __device__ __forceinline__ unsigned f2h_bits(float f) {
    _Float16 h = (_Float16)f;                       // v_cvt_f16_f32
    return (unsigned)__builtin_bit_cast(unsigned short, h);
}

// ---------------------------------------------------------------------------
// Kernel 0: out[b][o] = mean_c b[c][o]  (bias term; GEMM atomics add on top)
// ---------------------------------------------------------------------------
__global__ __launch_bounds__(256) void bias_init_kernel(const float* __restrict__ bias,
                                                        float* __restrict__ out) {
    const int o = threadIdx.x;                       // 0..255, coalesced over lanes
    float s = 0.f;
    for (int c = 0; c < CC; ++c) s += bias[(size_t)c * OO + o];
    s *= (1.0f / (float)CC);
    for (int b = 0; b < BB; ++b) out[(size_t)b * OO + o] = s;
}

// ---------------------------------------------------------------------------
// Kernel 1: A[b][c*720+t] = (x[b][t][c] - segmean) * (1/C), packed to fp16.
// One thread per (b, seg, c). Reads coalesced over c; writes 6x b128 per lane.
// ---------------------------------------------------------------------------
__global__ __launch_bounds__(256) void pack_kernel(const float* __restrict__ x,
                                                   _Float16* __restrict__ A) {
    const int idx = blockIdx.x * 256 + threadIdx.x;  // exactly B*NSEG*C threads
    const int c   = idx % CC;
    const int r   = idx / CC;
    const int seg = r % NSEG;
    const int b   = r / NSEG;

    const float* xp = x + ((size_t)b * TT + (size_t)seg * SEG) * CC + c;
    float v[SEG];
    float s = 0.f;
#pragma unroll
    for (int t = 0; t < SEG; ++t) { v[t] = xp[(size_t)t * CC]; s += v[t]; }
    const float mean  = s * (1.0f / (float)SEG);
    const float scale = 1.0f / (float)CC;            // fold channel-mean into A

    _Float16* ap = A + (size_t)b * KK + (size_t)c * TT + (size_t)seg * SEG;
#pragma unroll
    for (int j = 0; j < 6; ++j) {                    // 6 x 16B stores = 48 halves
        uint4 u;
        unsigned w[4];
#pragma unroll
        for (int q = 0; q < 4; ++q) {
            const int t = j * 8 + q * 2;
            w[q] = f2h_bits((v[t] - mean) * scale) |
                   (f2h_bits((v[t + 1] - mean) * scale) << 16);
        }
        u.x = w[0]; u.y = w[1]; u.z = w[2]; u.w = w[3];
        *(uint4*)(ap + j * 8) = u;
    }
}

// ---------------------------------------------------------------------------
// Kernel 2: split-K GEMM  out += A[128 x K] * W[K x 256]  via WMMA fp16->f32.
// block = 256 threads (8 waves). Each wave owns one 16-row M tile x 64 N cols.
// grid = (NBLK=4 n-blocks, KSPLIT k-splits); chunks grid-strided over K.
// ---------------------------------------------------------------------------
__global__ __launch_bounds__(256) void gemm_kernel(const _Float16* __restrict__ A,
                                                   const float* __restrict__ Wg,
                                                   float* __restrict__ out) {
    __shared__ __align__(32) _Float16 ldsA[BB * STRA];   // 26.6 KB
    __shared__ __align__(32) _Float16 ldsW[64 * STRW];   // 14.3 KB, transposed [n][k]

    const int tid   = threadIdx.x;
    const int lane  = tid & 31;
    const int wave  = tid >> 5;
    const int nbase = blockIdx.x * 64;
    const int m0    = wave * 16;
    const int lrow  = lane & 15;                     // M row / N col within fragment
    const bool loHalf = (lane < 16);

    // Per-thread A-staging geometry: each thread copies half a row (6 x 16B).
    const int arow  = tid >> 1;                      // 0..127
    const int acol  = (tid & 1) * 48;                // halves: 0 or 48

    f8 acc[4] = {f8{}, f8{}, f8{}, f8{}};

    for (int chunk = blockIdx.y; chunk < NC; chunk += gridDim.y) {
        const int k0 = chunk * KC;

        // ---- stage A chunk: 128 rows x 96 halves, async global->LDS ----
        {
            const _Float16* gsrc = A + (size_t)arow * KK + k0 + acol;
            _Float16*       ldst = &ldsA[arow * STRA + acol];
#pragma unroll
            for (int j = 0; j < 6; ++j)
                copy16_to_lds(gsrc + j * 8, ldst + j * 8);
        }

        // ---- stage W chunk transposed: ldsW[n][k] fp16, convert f32->f16 ----
        {
            const int nl = tid & 63;
            const int kq = (tid >> 6) * 2;           // {0,2,4,6}
#pragma unroll
            for (int p = 0; p < 12; ++p) {
                const int kk = p * 8 + kq;           // even k in [0,96)
                const float* wp = Wg + ((size_t)(k0 + kk)) * OO + nbase + nl;
                const unsigned u = f2h_bits(wp[0]) | (f2h_bits(wp[OO]) << 16);
                *(unsigned*)(&ldsW[nl * STRW + kk]) = u;
            }
        }

        // ---- prefetch next chunk (gfx1250 global_prefetch_b8) ----
        const int nchunk = chunk + gridDim.y;
        if (nchunk < NC) {
            const int rw = tid >> 1, hw = tid & 1;
            if (rw < KC)   // W: 96 rows x 256B
                __builtin_prefetch(Wg + ((size_t)(nchunk * KC + rw)) * OO + nbase + hw * 32, 0, 1);
            __builtin_prefetch(A + (size_t)(tid >> 1) * KK + nchunk * KC + (tid & 1) * 64, 0, 1);
        }

        wait_async_copies();
        __syncthreads();

        // ---- compute: 3 k-steps of 32, 4 N tiles each ----
#pragma unroll
        for (int ks = 0; ks < 3; ++ks) {
            const int kb   = ks * 32;
            const int aoff = kb + (loHalf ? 0 : 16); // A frag: contiguous 16 K {aoff..+15}
            h8 alo = *(const h8*)(&ldsA[(m0 + lrow) * STRA + aoff]);
            h8 ahi = *(const h8*)(&ldsA[(m0 + lrow) * STRA + aoff + 8]);
            h16 afrag = __builtin_shufflevector(alo, ahi,
                0, 1, 2, 3, 4, 5, 6, 7, 8, 9, 10, 11, 12, 13, 14, 15);
            const int boff = kb + (loHalf ? 0 : 16); // B frag: contiguous 16 K
#pragma unroll
            for (int nt = 0; nt < 4; ++nt) {
                h16 bfrag = *(const h16*)(&ldsW[(nt * 16 + lrow) * STRW + boff]);
                acc[nt] = __builtin_amdgcn_wmma_f32_16x16x32_f16(
                    false, afrag, false, bfrag, (short)0, acc[nt], false, false);
            }
        }
        __syncthreads();
    }

    // ---- reduce split-K partials into out via f32 atomics ----
    const int mrow = m0 + (loHalf ? 0 : 8);
#pragma unroll
    for (int nt = 0; nt < 4; ++nt) {
        const int n = nbase + nt * 16 + lrow;
#pragma unroll
        for (int r = 0; r < 8; ++r) {
            atomicAdd(&out[(size_t)(mrow + r) * OO + n], acc[nt][r]);
        }
    }
}

// ---------------------------------------------------------------------------
extern "C" void kernel_launch(void* const* d_in, const int* in_sizes, int n_in,
                              void* d_out, int out_size, void* d_ws, size_t ws_size,
                              hipStream_t stream) {
    const float* x    = (const float*)d_in[0];   // [128,720,862] f32
    const float* Wg   = (const float*)d_in[1];   // [862,720,256] f32 == [K,256] row-major
    const float* bias = (const float*)d_in[2];   // [862,256] f32
    float* out        = (float*)d_out;           // [128,256] f32
    _Float16* A       = (_Float16*)d_ws;         // packed normalized x: 128*620640 fp16 (~159MB)

    bias_init_kernel<<<1, 256, 0, stream>>>(bias, out);
    pack_kernel<<<(BB * NSEG * CC) / 256, 256, 0, stream>>>(x, A);   // exactly 6465 blocks
    gemm_kernel<<<dim3(NBLK, KSPLIT), 256, 0, stream>>>(A, Wg, out);
}
